// EquivariantGNN_e3nn_44736379355273
// MI455X (gfx1250) — matrix operand order repaired and checked by reading, not compiled, not run on hardware
//
#include <hip/hip_runtime.h>
#include <hip/hip_bf16.h>
#include <math.h>

// ---------------------------------------------------------------------------
// MI455X / gfx1250 implementation of the equivariant GNN forward pass.
//
// Design notes (compile-only environment; reasoning from MI455X specs):
//  * Total FLOPs dominated by the per-edge radial MLP GEMM (B*E,32)@(32,648)
//    ~= 10.9 GFLOP over both blocks; all intermediates (<40MB) fit in the
//    192MB L2, so the kernel is latency/atomic bound, not HBM bound.
//  * Radial GEMM runs on the matrix pipes: v_wmma_f32_16x16x32_bf16 -- the
//    K=32 of one WMMA exactly matches the radial hidden width, so a single
//    WMMA produces a 16-edge x 16-col f32 weight tile.
//  * The B operand (w2) is pre-swizzled into WMMA fragment order so each
//    lane fetches its 16 bf16 elements with one contiguous 32B load
//    (2x global_load_b128) instead of 16 strided u16 loads.
//  * The D tile is written to LDS in [col][edge] order so each lane's 8
//    accumulators are contiguous (2x ds_store_b128 instead of 8x b32).
//  * The GEMM is fused into the edge kernel so the 648 weights/edge
//    (340MB/block if materialized) never leave LDS.
//  * Scatter-add uses hardware global_atomic_add_f32 (agent scope, relaxed).
//
// Parameter flattening assumption: the harness flattens the params dict the
// jax way (sorted keys at each dict level, lists in order):
//   7..18  blocks[0]: lin_in{b,w_0e,w_1e,w_1o,w_2e} lin_res{...} radial{w1,w2}
//   19..30 blocks[1]: same
//   31..35 enc: b, w_0e, w_1e, w_1o, w_2e
//   36..38 to_sym: b, w_0e, w_2e
// ---------------------------------------------------------------------------

typedef __attribute__((ext_vector_type(16))) __bf16 v16bf;
typedef __attribute__((ext_vector_type(8)))  float  v8f;
typedef __attribute__((ext_vector_type(4)))  float  v4fx;

#define B_   4
#define N_   4096
#define E_   32768
#define BN   (B_ * N_)
#define BE   (B_ * E_)
#define HD   66           // hidden irreps dim: 12*1 + 4*3 + 4*3 + 6*5
#define NW   648          // TP weight count per edge
#define NWP  656          // padded to 41 tiles of 16
#define NTILE 41
#define RH   32           // radial hidden width (== WMMA K)

// ---- CG (Wigner-3j) flat-table offsets -------------------------------------
#define CG000 0
#define CG011 1
#define CG022 10
#define CG101 35
#define CG110 44
#define CG111 53
#define CG112 80
#define CG121 125
#define CG122 170
#define CG202 245
#define CG211 270
#define CG220 315
#define CG221 340
#define CG222 415
#define CGTOT 540

// ---- TP instruction metadata (17 instructions, reference iteration order) --
__constant__ int   TPI_X1OFF[17] = {0,0,0, 12,12,12,12,12, 24,24,24,24, 36,36,36,36,36};
__constant__ int   TPI_M1[17]    = {12,12,12, 4,4,4,4,4, 4,4,4,4, 6,6,6,6,6};
__constant__ int   TPI_NI[17]    = {1,1,1, 3,3,3,3,3, 3,3,3,3, 5,5,5,5,5};
__constant__ int   TPI_SHOFF[17] = {0,1,4, 0,1,1,1,4, 0,1,4,4, 0,1,4,4,4};
__constant__ int   TPI_NJ[17]    = {1,3,5, 1,3,3,3,5, 1,3,5,5, 1,3,5,5,5};
__constant__ int   TPI_WOFF[17]  = {0,144,192, 264,280,328,344,368, 384,400,416,432, 456,492,516,588,612};
__constant__ int   TPI_CGOFF[17] = {CG000,CG011,CG022, CG101,CG110,CG111,CG112,CG121,
                                    CG101,CG111,CG121,CG122, CG202,CG211,CG220,CG221,CG222};
__constant__ int   TPI_OUTOFF[17]= {0,12,36, 12,0,24,36,12, 24,12,24,36, 36,12,0,24,36};
__constant__ int   TPI_MO[17]    = {12,4,6, 4,12,4,6,4, 4,4,4,6, 6,4,12,4,6};
__constant__ int   TPI_NK[17]    = {1,3,5, 3,1,3,5,3, 3,3,3,5, 5,3,1,3,5};
__constant__ float TPI_ALPHA[17] = {
  0.31622776f, 0.31622776f, 0.39528471f,
  0.31622776f, 0.31622776f, 0.40824829f, 0.39528471f, 0.31622776f,
  0.40824829f, 0.31622776f, 0.40824829f, 0.39528471f,
  0.39528471f, 0.31622776f, 0.31622776f, 0.40824829f, 0.39528471f };

__device__ __forceinline__ float sigm(float x) { return 1.0f / (1.0f + expf(-x)); }
__device__ __forceinline__ float epsf(int i, int j, int k) {
  return (float)((i - j) * (j - k) * (k - i)) * 0.5f;
}

// ---------------------------------------------------------------------------
// CG table builder (closed forms in the reference's real-SH basis, each table
// L2-normalized like the reference's SVD null-vector construction).
// ---------------------------------------------------------------------------
__global__ void init_cg_kernel(float* __restrict__ cg) {
  if (blockIdx.x != 0 || threadIdx.x != 0) return;
  for (int i = 0; i < CGTOT; ++i) cg[i] = 0.0f;
  const float is2 = 0.70710678f, is3 = 0.57735027f, is5 = 0.44721360f, is6 = 0.40824829f;

  // Symmetric-traceless basis matching _sh2_base order: [xz, xy, y2-.., yz, z2-x2]
  float M[5][3][3] = {};
  M[0][0][2] = is2;  M[0][2][0] = is2;
  M[1][0][1] = is2;  M[1][1][0] = is2;
  M[2][0][0] = -is6; M[2][1][1] = 2.0f * is6; M[2][2][2] = -is6;
  M[3][1][2] = is2;  M[3][2][1] = is2;
  M[4][0][0] = -is2; M[4][2][2] = is2;

  cg[CG000] = 1.0f;
  for (int j = 0; j < 3; ++j) cg[CG011 + 4 * j] = is3;
  for (int a = 0; a < 5; ++a) cg[CG022 + 6 * a] = is5;
  for (int i = 0; i < 3; ++i) cg[CG101 + 4 * i] = is3;
  for (int i = 0; i < 3; ++i) cg[CG110 + 4 * i] = is3;
  for (int i = 0; i < 3; ++i)
    for (int j = 0; j < 3; ++j)
      for (int k = 0; k < 3; ++k)
        cg[CG111 + (i * 3 + j) * 3 + k] = epsf(i, j, k) * is6;
  for (int i = 0; i < 3; ++i)
    for (int j = 0; j < 3; ++j)
      for (int a = 0; a < 5; ++a)
        cg[CG112 + (i * 3 + j) * 5 + a] = M[a][i][j] * is5;
  for (int i = 0; i < 3; ++i)
    for (int a = 0; a < 5; ++a)
      for (int k = 0; k < 3; ++k)
        cg[CG121 + (i * 5 + a) * 3 + k] = M[a][i][k] * is5;
  for (int a = 0; a < 5; ++a) cg[CG202 + 6 * a] = is5;
  for (int a = 0; a < 5; ++a)
    for (int j = 0; j < 3; ++j)
      for (int k = 0; k < 3; ++k)
        cg[CG211 + (a * 3 + j) * 3 + k] = M[a][j][k] * is5;
  for (int a = 0; a < 5; ++a) cg[CG220 + 6 * a] = is5;

  // (1,2,2): t[i][a][b] = eps_{ipq} (Ma Mb)_{pq}, normalized
  {
    float nrm = 0.0f;
    for (int i = 0; i < 3; ++i)
      for (int a = 0; a < 5; ++a)
        for (int b = 0; b < 5; ++b) {
          float v = 0.0f;
          for (int p = 0; p < 3; ++p)
            for (int q = 0; q < 3; ++q) {
              float pq = 0.0f;
              for (int r = 0; r < 3; ++r) pq += M[a][p][r] * M[b][r][q];
              v += epsf(i, p, q) * pq;
            }
          cg[CG122 + (i * 5 + a) * 5 + b] = v;
          nrm += v * v;
        }
    float s = (nrm > 0.0f) ? rsqrtf(nrm) : 0.0f;
    for (int q = 0; q < 75; ++q) cg[CG122 + q] *= s;
  }
  // (2,2,1): t[a][b][k] = eps_{kpq} (Ma Mb)_{pq}, normalized
  {
    float nrm = 0.0f;
    for (int a = 0; a < 5; ++a)
      for (int b = 0; b < 5; ++b)
        for (int k = 0; k < 3; ++k) {
          float v = 0.0f;
          for (int p = 0; p < 3; ++p)
            for (int q = 0; q < 3; ++q) {
              float pq = 0.0f;
              for (int r = 0; r < 3; ++r) pq += M[a][p][r] * M[b][r][q];
              v += epsf(k, p, q) * pq;
            }
          cg[CG221 + (a * 5 + b) * 3 + k] = v;
          nrm += v * v;
        }
    float s = (nrm > 0.0f) ? rsqrtf(nrm) : 0.0f;
    for (int q = 0; q < 75; ++q) cg[CG221 + q] *= s;
  }
  // (2,2,2): t[a][b][c] = tr(MaMbMc) + tr(MaMcMb), normalized
  {
    float nrm = 0.0f;
    for (int a = 0; a < 5; ++a)
      for (int b = 0; b < 5; ++b)
        for (int c = 0; c < 5; ++c) {
          float v = 0.0f;
          for (int p = 0; p < 3; ++p)
            for (int q = 0; q < 3; ++q)
              for (int r = 0; r < 3; ++r)
                v += M[a][p][q] * (M[b][q][r] * M[c][r][p] + M[c][q][r] * M[b][r][p]);
          cg[CG222 + (a * 5 + b) * 5 + c] = v;
          nrm += v * v;
        }
    float s = (nrm > 0.0f) ? rsqrtf(nrm) : 0.0f;
    for (int q = 0; q < 125; ++q) cg[CG222 + q] *= s;
  }
}

// ---------------------------------------------------------------------------
// Node encoder: x15 -> irreps15 -> HID (66)
// ---------------------------------------------------------------------------
__global__ __launch_bounds__(256) void encode_kernel(
    const float* __restrict__ xn, const float* __restrict__ mask,
    const float* __restrict__ b0, const float* __restrict__ w0,
    const float* __restrict__ w1e, const float* __restrict__ w1o,
    const float* __restrict__ w2, float* __restrict__ h) {
  int n = blockIdx.x * blockDim.x + threadIdx.x;
  if (n >= BN) return;
  const float* x = xn + (size_t)n * 15;
  float m = mask[n];
  float p[3] = {x[0], x[1], x[2]};
  float t[3] = {x[3], x[4], x[5]};
  float U[3][3];
#pragma unroll
  for (int r = 0; r < 3; ++r)
#pragma unroll
    for (int c = 0; c < 3; ++c) U[r][c] = x[6 + 3 * r + c];
  float S01 = 0.5f * (U[0][1] + U[1][0]);
  float S02 = 0.5f * (U[0][2] + U[2][0]);
  float S12 = 0.5f * (U[1][2] + U[2][1]);
  float tr = U[0][0] + U[1][1] + U[2][2];
  const float is6 = 0.40824829f, s2 = 1.41421356f;
  float a[5];
  a[0] = (2.0f * U[2][2] - U[0][0] - U[1][1]) * is6;
  a[1] = (U[0][0] - U[1][1]) * 0.70710678f;
  a[2] = s2 * S01;
  a[3] = s2 * S02;
  a[4] = s2 * S12;
  float om[3] = {0.5f * (U[1][2] - U[2][1]),
                 0.5f * (U[2][0] - U[0][2]),
                 0.5f * (U[0][1] - U[1][0])};
  float* ho = h + (size_t)n * HD;
#pragma unroll
  for (int c = 0; c < 12; ++c) ho[c] = (tr * w0[c] + b0[c]) * m;
#pragma unroll
  for (int u = 0; u < 4; ++u)
#pragma unroll
    for (int i = 0; i < 3; ++i)
      ho[12 + u * 3 + i] = (p[i] * w1o[u] + t[i] * w1o[4 + u]) * 0.70710678f * m;
#pragma unroll
  for (int u = 0; u < 4; ++u)
#pragma unroll
    for (int i = 0; i < 3; ++i) ho[24 + u * 3 + i] = om[i] * w1e[u] * m;
#pragma unroll
  for (int u = 0; u < 6; ++u)
#pragma unroll
    for (int k = 0; k < 5; ++k) ho[36 + u * 5 + k] = a[k] * w2[u] * m;
}

// ---------------------------------------------------------------------------
// Fused HID->HID linear (lin_in and lin_res in one pass)
// ---------------------------------------------------------------------------
__global__ __launch_bounds__(256) void node_linear_kernel(
    const float* __restrict__ h,
    const float* __restrict__ bA, const float* __restrict__ w0A,
    const float* __restrict__ w1eA, const float* __restrict__ w1oA,
    const float* __restrict__ w2A,
    const float* __restrict__ bR, const float* __restrict__ w0R,
    const float* __restrict__ w1eR, const float* __restrict__ w1oR,
    const float* __restrict__ w2R,
    float* __restrict__ hin, float* __restrict__ res) {
  int n = blockIdx.x * blockDim.x + threadIdx.x;
  if (n >= BN) return;
  float x[HD];
#pragma unroll
  for (int c = 0; c < HD; ++c) x[c] = h[(size_t)n * HD + c];
  float* oA = hin + (size_t)n * HD;
  float* oR = res + (size_t)n * HD;
  const float i12 = 0.28867513f, i6 = 0.40824829f;
  for (int c = 0; c < 12; ++c) {
    float sA = 0.0f, sR = 0.0f;
#pragma unroll
    for (int u = 0; u < 12; ++u) {
      sA += x[u] * w0A[u * 12 + c];
      sR += x[u] * w0R[u * 12 + c];
    }
    oA[c] = sA * i12 + bA[c];
    oR[c] = sR * i12 + bR[c];
  }
  for (int c = 0; c < 4; ++c)
    for (int i = 0; i < 3; ++i) {
      float sA = 0.0f, sR = 0.0f, sAe = 0.0f, sRe = 0.0f;
#pragma unroll
      for (int u = 0; u < 4; ++u) {
        float vo = x[12 + u * 3 + i], ve = x[24 + u * 3 + i];
        sA += vo * w1oA[u * 4 + c];  sR += vo * w1oR[u * 4 + c];
        sAe += ve * w1eA[u * 4 + c]; sRe += ve * w1eR[u * 4 + c];
      }
      oA[12 + c * 3 + i] = sA * 0.5f;  oR[12 + c * 3 + i] = sR * 0.5f;
      oA[24 + c * 3 + i] = sAe * 0.5f; oR[24 + c * 3 + i] = sRe * 0.5f;
    }
  for (int c = 0; c < 6; ++c)
    for (int k = 0; k < 5; ++k) {
      float sA = 0.0f, sR = 0.0f;
#pragma unroll
      for (int u = 0; u < 6; ++u) {
        float v = x[36 + u * 5 + k];
        sA += v * w2A[u * 6 + c];
        sR += v * w2R[u * 6 + c];
      }
      oA[36 + c * 5 + k] = sA * i6;
      oR[36 + c * 5 + k] = sR * i6;
    }
}

// ---------------------------------------------------------------------------
// w2 (32x648 f32) -> bf16 in WMMA B-fragment order:
//   out[((t*32 + lane)*16) + 2*v + p] = w2[k][col]
//   where half=lane>>4, r=lane&15, k = half*16 + 2*v + p, col = t*16 + r.
// Each lane's fragment for tile t is then one contiguous 32B load.
// ---------------------------------------------------------------------------
__global__ __launch_bounds__(256) void w2cvt_kernel(const float* __restrict__ w2,
                                                    __bf16* __restrict__ w2b) {
  int idx = blockIdx.x * blockDim.x + threadIdx.x;
  if (idx >= NTILE * 32 * 16) return;
  int e    = idx & 15;
  int lane = (idx >> 4) & 31;
  int t    = idx >> 9;
  int half = lane >> 4, r = lane & 15;
  int k   = half * 16 + e;          // e == 2*v + p, contiguous K within half
  int col = t * 16 + r;
  float v = (col < NW) ? w2[k * NW + col] : 0.0f;
  w2b[idx] = (__bf16)v;
}

// ---------------------------------------------------------------------------
// Fused edge kernel: SH + radial hidden -> WMMA radial GEMM (bf16, f32 acc)
// -> tensor product -> scatter-add. 16 edges / workgroup, 4 waves.
// ---------------------------------------------------------------------------
__global__ __launch_bounds__(128) void radial_tp_kernel(
    const int* __restrict__ e_src, const int* __restrict__ e_dst,
    const float* __restrict__ e_attr, const float* __restrict__ mask_edges,
    const float* __restrict__ w1, const __bf16* __restrict__ w2b,
    const float* __restrict__ hin, const float* __restrict__ cgb,
    float* __restrict__ agg) {
  __shared__ float  wls[NWP][16];   // per-edge TP weights, [col][edge] (D tiles)
  __shared__ __bf16 h1s[16][RH];    // radial hidden, bf16 (WMMA A)
  __shared__ float  Ys[16][9];
  __shared__ float  rns[16];
  __shared__ float  xs[16][HD];     // gathered h_in[e_src]
  __shared__ float  msgs[16][HD];
  __shared__ int    dsts[16];
  __shared__ int    srcs[16];

  const int tid = threadIdx.x;
  const int base = blockIdx.x * 16;

  if (tid < 16) {
    int ge = base + tid;
    int b = ge >> 15;                 // E_ == 32768
    float m = mask_edges[ge];
    float ex = e_attr[3 * ge + 0] * m;
    float ey = e_attr[3 * ge + 1] * m;
    float ez = e_attr[3 * ge + 2] * m;
    float rn = sqrtf(ex * ex + ey * ey + ez * ez) + 1e-8f;
    rns[tid] = rn;
    float inv = 1.0f / rn;
    float x = ex * inv, y = ey * inv, z = ez * inv;
    const float s3 = 1.73205081f, s5 = 2.23606798f, s15 = 3.87298335f;
    Ys[tid][0] = 1.0f;
    Ys[tid][1] = s3 * x; Ys[tid][2] = s3 * y; Ys[tid][3] = s3 * z;
    Ys[tid][4] = s15 * x * z;
    Ys[tid][5] = s15 * x * y;
    Ys[tid][6] = s5 * (y * y - 0.5f * (x * x + z * z));
    Ys[tid][7] = s15 * y * z;
    Ys[tid][8] = 0.5f * s15 * (z * z - x * x);
    srcs[tid] = b * N_ + e_src[ge];
    dsts[tid] = b * N_ + e_dst[ge];
  }
  if (tid < NTILE) __builtin_prefetch(w2b + tid * 512, 0, 3);  // global_prefetch_b8
  __syncthreads();

  for (int idx = tid; idx < 16 * RH; idx += 128) {
    int i = idx >> 5, j = idx & 31;
    float pre = rns[i] * w1[j];
    float sil = pre * sigm(pre);
    // fold SILU_C and 1/sqrt(32) into the A operand
    h1s[i][j] = (__bf16)(1.679177f * sil * 0.17677669529f);
  }
  for (int idx = tid; idx < 16 * HD; idx += 128) {
    int i = idx / HD, c = idx - i * HD;
    xs[i][c] = hin[(size_t)srcs[i] * HD + c];
    msgs[i][c] = 0.0f;
  }
  __syncthreads();

  // ---- WMMA radial GEMM: (16 edges x 32) @ (32 x 656) ----------------------
  const int wave = tid >> 5, lane = tid & 31;
  const int half = lane >> 4, r = lane & 15;
  v16bf afrag;
#pragma unroll
  for (int v = 0; v < 8; ++v) {
    // A as packed here (contiguous K, same as w2b): lanes 0-15 hold K{0..15}, lanes 16-31 K{16..31}
    int k = half * 16 + 2 * v;
    afrag[2 * v]     = h1s[r][k];
    afrag[2 * v + 1] = h1s[r][k + 1];
  }
  const v16bf* w2f = (const v16bf*)w2b;   // fragment-order B operand
  for (int t = wave; t < NTILE; t += 4) {
    v16bf bfrag = w2f[t * 32 + lane];     // one 32B contiguous load per lane
    v8f c = {};
    c = __builtin_amdgcn_wmma_f32_16x16x32_bf16(false, afrag, false, bfrag,
                                                (short)0, c, false, false);
    // D layout: VGPR rr -> row rr (lanes 0-15) / rr+8 (lanes 16-31), col = lane&15.
    // Transposed LDS tile makes the 8 rows contiguous: 2x ds_store_b128.
    int col = t * 16 + r;
    v4fx lo = {c[0], c[1], c[2], c[3]};
    v4fx hi = {c[4], c[5], c[6], c[7]};
    *(v4fx*)&wls[col][8 * half]     = lo;
    *(v4fx*)&wls[col][8 * half + 4] = hi;
  }
  __syncthreads();

  // ---- Tensor product: 16 edges x 17 instructions over 128 threads --------
  for (int task = tid; task < 16 * 17; task += 128) {
    int i = task / 17, s = task - i * 17;
    const int m1 = TPI_M1[s], ni = TPI_NI[s], nj = TPI_NJ[s];
    const int mo = TPI_MO[s], nk = TPI_NK[s];
    const float* x1 = &xs[i][TPI_X1OFF[s]];
    const float* Yv = &Ys[i][TPI_SHOFF[s]];
    const float* cg = cgb + TPI_CGOFF[s];
    const float* wv = &wls[TPI_WOFF[s]][i];   // stride 16 floats per weight index
    float* outp = &msgs[i][TPI_OUTOFF[s]];
    const float al = TPI_ALPHA[s];
    float acc[30];  // mo*nk <= 30
    for (int q = 0; q < mo * nk; ++q) acc[q] = 0.0f;
    for (int u = 0; u < m1; ++u) {
      float z[5] = {0.0f, 0.0f, 0.0f, 0.0f, 0.0f};
      for (int ii = 0; ii < ni; ++ii) {
        float xv = x1[u * ni + ii];
        for (int j = 0; j < nj; ++j) {
          float f = xv * Yv[j];
          const float* cgr = cg + (ii * nj + j) * nk;
          for (int k = 0; k < nk; ++k) z[k] += f * cgr[k];
        }
      }
      const float* wu = wv + (size_t)(u * mo) * 16;
      for (int w = 0; w < mo; ++w) {
        float wt = wu[w * 16];
        for (int k = 0; k < nk; ++k) acc[w * nk + k] += wt * z[k];
      }
    }
    for (int q = 0; q < mo * nk; ++q)
      __hip_atomic_fetch_add(&outp[q], al * acc[q], __ATOMIC_RELAXED,
                             __HIP_MEMORY_SCOPE_WORKGROUP);  // ds_add_f32
  }
  __syncthreads();

  // ---- Scatter-add to destination nodes ------------------------------------
  for (int idx = tid; idx < 16 * HD; idx += 128) {
    int i = idx / HD, c = idx - i * HD;
    __hip_atomic_fetch_add(&agg[(size_t)dsts[i] * HD + c], msgs[i][c],
                           __ATOMIC_RELAXED, __HIP_MEMORY_SCOPE_AGENT);
  }
}

// ---------------------------------------------------------------------------
// h = norm_act(agg + res) * mask     (silu(n)/n == sigmoid(n))
// ---------------------------------------------------------------------------
__global__ __launch_bounds__(256) void norm_act_kernel(
    const float* __restrict__ agg, const float* __restrict__ res,
    const float* __restrict__ mask, float* __restrict__ h) {
  int n = blockIdx.x * blockDim.x + threadIdx.x;
  if (n >= BN) return;
  float m = mask[n];
  float x[HD];
#pragma unroll
  for (int c = 0; c < HD; ++c)
    x[c] = agg[(size_t)n * HD + c] + res[(size_t)n * HD + c];
  float* o = h + (size_t)n * HD;
#pragma unroll
  for (int c = 0; c < 12; ++c) {
    float f = x[c];
    float nn = sqrtf(f * f + 1e-12f);
    o[c] = f * sigm(nn) * m;
  }
#pragma unroll
  for (int u = 0; u < 8; ++u) {   // 4x 1o then 4x 1e
    int base = 12 + u * 3;
    float s = x[base] * x[base] + x[base + 1] * x[base + 1] + x[base + 2] * x[base + 2];
    float fac = sigm(sqrtf(s + 1e-12f)) * m;
    o[base] = x[base] * fac; o[base + 1] = x[base + 1] * fac; o[base + 2] = x[base + 2] * fac;
  }
#pragma unroll
  for (int u = 0; u < 6; ++u) {
    int base = 36 + u * 5;
    float s = 1e-12f;
#pragma unroll
    for (int k = 0; k < 5; ++k) s += x[base + k] * x[base + k];
    float fac = sigm(sqrtf(s)) * m;
#pragma unroll
    for (int k = 0; k < 5; ++k) o[base + k] = x[base + k] * fac;
  }
}

// ---------------------------------------------------------------------------
// to_sym at center nodes + Mandel
// ---------------------------------------------------------------------------
__global__ void output_kernel(const float* __restrict__ h,
                              const int* __restrict__ centers,
                              const float* __restrict__ sb,
                              const float* __restrict__ sw0,
                              const float* __restrict__ sw2,
                              float* __restrict__ out) {
  int b = threadIdx.x;
  if (b >= B_) return;
  const float* x = h + ((size_t)b * N_ + centers[b]) * HD;
  float s0 = 0.0f;
#pragma unroll
  for (int u = 0; u < 12; ++u) s0 += x[u] * sw0[u];
  s0 = s0 * 0.28867513f + sb[0];
  float s2[5];
#pragma unroll
  for (int k = 0; k < 5; ++k) {
    float a = 0.0f;
#pragma unroll
    for (int u = 0; u < 6; ++u) a += x[36 + u * 5 + k] * sw2[u];
    s2[k] = a * 0.40824829f;
  }
  float tr = s0, a0 = s2[0], a2c = s2[1], a2s = s2[2], a1c = s2[3], a1s = s2[4];
  const float is6 = 0.40824829f, is2 = 0.70710678f;
  float Sxx = -a0 * is6 + a2c * is2 + tr * (1.0f / 3.0f);
  float Syy = -a0 * is6 - a2c * is2 + tr * (1.0f / 3.0f);
  float Szz = 2.0f * a0 * is6 + tr * (1.0f / 3.0f);
  out[b * 6 + 0] = Sxx;
  out[b * 6 + 1] = a2s;
  out[b * 6 + 2] = a1c;
  out[b * 6 + 3] = Syy;
  out[b * 6 + 4] = a1s;
  out[b * 6 + 5] = Szz;
}

// ---------------------------------------------------------------------------
extern "C" void kernel_launch(void* const* d_in, const int* in_sizes, int n_in,
                              void* d_out, int out_size, void* d_ws, size_t ws_size,
                              hipStream_t stream) {
  (void)in_sizes; (void)n_in; (void)out_size; (void)ws_size;
  const float* x_nodes    = (const float*)d_in[0];
  const int*   e_src      = (const int*)d_in[1];
  const int*   e_dst      = (const int*)d_in[2];
  const float* e_attr     = (const float*)d_in[3];
  const float* mask_nodes = (const float*)d_in[4];
  const float* mask_edges = (const float*)d_in[5];
  const int*   centers    = (const int*)d_in[6];

  const float* li_b[2]  = {(const float*)d_in[7],  (const float*)d_in[19]};
  const float* li_w0[2] = {(const float*)d_in[8],  (const float*)d_in[20]};
  const float* li_w1e[2]= {(const float*)d_in[9],  (const float*)d_in[21]};
  const float* li_w1o[2]= {(const float*)d_in[10], (const float*)d_in[22]};
  const float* li_w2[2] = {(const float*)d_in[11], (const float*)d_in[23]};
  const float* lr_b[2]  = {(const float*)d_in[12], (const float*)d_in[24]};
  const float* lr_w0[2] = {(const float*)d_in[13], (const float*)d_in[25]};
  const float* lr_w1e[2]= {(const float*)d_in[14], (const float*)d_in[26]};
  const float* lr_w1o[2]= {(const float*)d_in[15], (const float*)d_in[27]};
  const float* lr_w2[2] = {(const float*)d_in[16], (const float*)d_in[28]};
  const float* rad_w1[2]= {(const float*)d_in[17], (const float*)d_in[29]};
  const float* rad_w2[2]= {(const float*)d_in[18], (const float*)d_in[30]};
  const float* enc_b   = (const float*)d_in[31];
  const float* enc_w0  = (const float*)d_in[32];
  const float* enc_w1e = (const float*)d_in[33];
  const float* enc_w1o = (const float*)d_in[34];
  const float* enc_w2  = (const float*)d_in[35];
  const float* sym_b   = (const float*)d_in[36];
  const float* sym_w0  = (const float*)d_in[37];
  const float* sym_w2  = (const float*)d_in[38];

  // workspace layout
  float*  H   = (float*)d_ws;
  float*  HIN = H   + (size_t)BN * HD;
  float*  RES = HIN + (size_t)BN * HD;
  float*  AGG = RES + (size_t)BN * HD;
  float*  CGB = AGG + (size_t)BN * HD;
  __bf16* W2B = (__bf16*)(CGB + 544);   // 41*32*16 bf16, 32B-aligned

  init_cg_kernel<<<1, 32, 0, stream>>>(CGB);
  encode_kernel<<<BN / 256, 256, 0, stream>>>(x_nodes, mask_nodes, enc_b, enc_w0,
                                              enc_w1e, enc_w1o, enc_w2, H);
  for (int blk = 0; blk < 2; ++blk) {
    w2cvt_kernel<<<(NTILE * 32 * 16 + 255) / 256, 256, 0, stream>>>(rad_w2[blk], W2B);
    node_linear_kernel<<<BN / 256, 256, 0, stream>>>(
        H, li_b[blk], li_w0[blk], li_w1e[blk], li_w1o[blk], li_w2[blk],
        lr_b[blk], lr_w0[blk], lr_w1e[blk], lr_w1o[blk], lr_w2[blk], HIN, RES);
    hipMemsetAsync(AGG, 0, (size_t)BN * HD * sizeof(float), stream);
    radial_tp_kernel<<<BE / 16, 128, 0, stream>>>(e_src, e_dst, e_attr, mask_edges,
                                                  rad_w1[blk], W2B, HIN, CGB, AGG);
    norm_act_kernel<<<BN / 256, 256, 0, stream>>>(AGG, RES, mask_nodes, H);
  }
  output_kernel<<<1, 32, 0, stream>>>(H, centers, sym_b, sym_w0, sym_w2, (float*)d_out);
}
